// Model_81870666596638
// MI455X (gfx1250) — matrix operand order repaired and checked by reading, not compiled, not run on hardware
//
#include <hip/hip_runtime.h>
#include <stdint.h>

// ---------------------------------------------------------------------------
// Types / helpers
// ---------------------------------------------------------------------------
typedef __attribute__((ext_vector_type(16))) __bf16 v16bf;
typedef __attribute__((ext_vector_type(8)))  float  v8f;

static __device__ __forceinline__ float bf2f(unsigned short u) {
  unsigned int v = ((unsigned int)u) << 16;
  return __builtin_bit_cast(float, v);
}
static __device__ __forceinline__ unsigned short f2bf(float f) {
  unsigned int u = __builtin_bit_cast(unsigned int, f);
  unsigned int r = u + 0x7FFFu + ((u >> 16) & 1u);   // round-to-nearest-even
  return (unsigned short)(r >> 16);
}
static __device__ __forceinline__ __bf16 us2bf(unsigned short u) {
  return __builtin_bit_cast(__bf16, u);
}

// gathered value: channels [0,C) = feat (bf16), [C,C+3) = xyz
static __device__ __forceinline__ float gval(const unsigned short* feat,
                                             const float* xyz,
                                             int b, int n, int c, int C, int N) {
  if (c < C) return bf2f(feat[((size_t)b * N + n) * C + c]);
  return xyz[((size_t)b * N + n) * 3 + (c - C)];
}

// ---------------------------------------------------------------------------
// WMMA bf16 GEMM:  Y[M,N](bf16) = X[M,K](bf16) * W[N,K]^T(bf16) + bias[N]
// Requires: M % 128 == 0, N % 128 == 0, K % 32 == 0 (buffers zero-padded).
// Block = 256 threads = 8 waves; block tile 128x128; wave tile 32x64
// (2x4 WMMA 16x16x32 tiles -> 8 v_wmma per K-step, 2 A-frags + 4 B-frags).
// ---------------------------------------------------------------------------
#define LSA 40
__global__ __launch_bounds__(256) void k_gemm(
    const unsigned short* __restrict__ X, int ldx,
    const unsigned short* __restrict__ W, int ldw,
    const float* __restrict__ bias,
    unsigned short* __restrict__ Y, int ldy,
    int M, int N, int K) {
  (void)M; (void)N;
  __shared__ unsigned short As[128 * LSA];
  __shared__ unsigned short Bs[128 * LSA];

  const int t    = threadIdx.x;
  const int lane = t & 31;
  const int wave = t >> 5;
  const int wy   = wave >> 1;          // 0..3 -> 32-row M sub-tile
  const int wx   = wave & 1;           // 0..1 -> 64-col N sub-tile
  const int m_blk = blockIdx.y * 128;
  const int n_blk = blockIdx.x * 128;

  v8f acc[2][4];
#pragma unroll
  for (int i = 0; i < 2; ++i)
#pragma unroll
    for (int j = 0; j < 4; ++j)
#pragma unroll
      for (int e = 0; e < 8; ++e) acc[i][j][e] = 0.0f;

  const int row = t >> 1;              // 0..127
  const int col = (t & 1) * 16;        // 0 or 16 (ushort units)

  for (int k0 = 0; k0 < K; k0 += 32) {
    const uint4* asrc = (const uint4*)(X + (size_t)(m_blk + row) * ldx + k0 + col);
    uint4 a0 = asrc[0];
    uint4 a1 = asrc[1];
    const uint4* bsrc = (const uint4*)(W + (size_t)(n_blk + row) * ldw + k0 + col);
    uint4 b0 = bsrc[0];
    uint4 b1 = bsrc[1];
    if (k0 + 32 < K) {
      __builtin_prefetch(X + (size_t)(m_blk + row) * ldx + k0 + 32 + col, 0, 1);
      __builtin_prefetch(W + (size_t)(n_blk + row) * ldw + k0 + 32 + col, 0, 1);
    }
    __syncthreads();                       // previous iteration's readers done
    *(uint4*)&As[row * LSA + col]     = a0;
    *(uint4*)&As[row * LSA + col + 8] = a1;
    *(uint4*)&Bs[row * LSA + col]     = b0;
    *(uint4*)&Bs[row * LSA + col + 8] = b1;
    __syncthreads();

    // A fragments (16x32 bf16: lanes 0-15 M=0..15 hold K 0..15;
    // lanes 16-31 same M rows hold K 16..31)
    const int half = lane >> 4;
    const int mr   = lane & 15;
    v16bf afr[2], bfr[4];
#pragma unroll
    for (int i = 0; i < 2; ++i) {
      const unsigned short* ap = &As[(wy * 32 + i * 16 + mr) * LSA];
#pragma unroll
      for (int j = 0; j < 8; ++j) {
        int kk = half * 16 + j * 2;
        afr[i][2 * j]     = us2bf(ap[kk]);
        afr[i][2 * j + 1] = us2bf(ap[kk + 1]);
      }
    }
    // B fragments (32x16 bf16: lanes 0-15 col N=lane K 0..15; lanes 16-31 K 16..31)
    const int nc = lane & 15;
    const int kh = lane >> 4;
#pragma unroll
    for (int j = 0; j < 4; ++j) {
      const unsigned short* bp = &Bs[(wx * 64 + j * 16 + nc) * LSA + kh * 16];
#pragma unroll
      for (int e = 0; e < 8; ++e) {
        bfr[j][2 * e]     = us2bf(bp[2 * e]);
        bfr[j][2 * e + 1] = us2bf(bp[2 * e + 1]);
      }
    }
#pragma unroll
    for (int i = 0; i < 2; ++i)
#pragma unroll
      for (int j = 0; j < 4; ++j)
        acc[i][j] = __builtin_amdgcn_wmma_f32_16x16x32_bf16(
            false, afr[i], false, bfr[j], (short)0, acc[i][j], false, false);
  }

  // C/D layout: VGPR r -> lanes 0-15: M=r, N=lane; lanes 16-31: M=r+8, N=lane-16
  const int mr2 = (lane < 16) ? 0 : 8;
  const int nc2 = lane & 15;
#pragma unroll
  for (int i = 0; i < 2; ++i)
#pragma unroll
    for (int j = 0; j < 4; ++j) {
      int colg = n_blk + wx * 64 + j * 16 + nc2;
      float bv = bias[colg];
#pragma unroll
      for (int r = 0; r < 8; ++r) {
        int rowg = m_blk + wy * 32 + i * 16 + mr2 + r;
        Y[(size_t)rowg * ldy + colg] = f2bf(acc[i][j][r] + bv);
      }
    }
}

// ---------------------------------------------------------------------------
// BatchNorm stats (training mode, biased var) over rows, per channel (bf16 in)
// ---------------------------------------------------------------------------
__global__ void k_bn_stats(const unsigned short* __restrict__ Y, int ld, int rows,
                           float* __restrict__ stats) {
  int c = blockIdx.x, t = threadIdx.x;
  float s = 0.f, q = 0.f;
  for (int r = t; r < rows; r += 256) {
    float v = bf2f(Y[(size_t)r * ld + c]);
    s += v; q += v * v;
  }
  __shared__ float ss[256], qq[256];
  ss[t] = s; qq[t] = q;
  __syncthreads();
  for (int k = 128; k; k >>= 1) {
    if (t < k) { ss[t] += ss[t + k]; qq[t] += qq[t + k]; }
    __syncthreads();
  }
  if (t == 0) {
    float m = ss[0] / rows;
    float var = qq[0] / rows - m * m;
    stats[2 * c]     = m;
    stats[2 * c + 1] = rsqrtf(var + 1e-5f);
  }
}

// apply: out = relu( g*(y-m)*rstd + be [+ residual] ); 8 bf16 per thread (b128)
__global__ void k_bnapply8(const unsigned short* __restrict__ Y,
                           const unsigned short* __restrict__ R,
                           unsigned short* __restrict__ O,
                           const float* __restrict__ stats,
                           const float* __restrict__ g,
                           const float* __restrict__ be,
                           size_t total8, int C) {
  size_t i = (size_t)blockIdx.x * 256 + threadIdx.x;
  if (i >= total8) return;
  size_t base = i * 8;
  int c0 = (int)(base % (size_t)C);        // C % 8 == 0 -> 8 contiguous channels
  uint4 yv = *(const uint4*)(Y + base);
  const unsigned short* yp = (const unsigned short*)&yv;
  uint4 rv = {0, 0, 0, 0};
  if (R) rv = *(const uint4*)(R + base);
  const unsigned short* rp = (const unsigned short*)&rv;
  uint4 ov;
  unsigned short* op = (unsigned short*)&ov;
#pragma unroll
  for (int e = 0; e < 8; ++e) {
    int c = c0 + e;
    float v = bf2f(yp[e]);
    v = g[c] * (v - stats[2 * c]) * stats[2 * c + 1] + be[c];
    if (R) v += bf2f(rp[e]);
    op[e] = f2bf(fmaxf(v, 0.f));
  }
  *(uint4*)(O + base) = ov;
}

// ---------------------------------------------------------------------------
// misc small kernels
// ---------------------------------------------------------------------------
__global__ void k_xpose(const float* __restrict__ x, float* __restrict__ xyz, int N) {
  int i = blockIdx.x * 256 + threadIdx.x;           // over B*N
  if (i >= 16 * N) return;
  int b = i / N, n = i % N;
  for (int d = 0; d < 3; ++d)
    xyz[(size_t)i * 3 + d] = x[((size_t)b * 3 + d) * N + n];
}

__global__ void k_convw(const float* __restrict__ w, unsigned short* __restrict__ o,
                        int cout, int cin, int ldo) {
  int i = blockIdx.x * 256 + threadIdx.x;
  if (i >= cout * ldo) return;
  int r = i / ldo, c = i % ldo;
  o[i] = (c < cin) ? f2bf(w[(size_t)r * cin + c]) : (unsigned short)0;
}

// small fp32 linear -> bf16 out (used for the 3->64 embedding)
__global__ void k_lin_small(const float* __restrict__ X, const float* __restrict__ W,
                            const float* __restrict__ b, unsigned short* __restrict__ Y,
                            int rows, int N, int K) {
  int i = blockIdx.x * 256 + threadIdx.x;
  if (i >= rows * N) return;
  int r = i / N, c = i % N;
  float s = b[c];
  for (int k = 0; k < K; ++k) s += X[(size_t)r * K + k] * W[(size_t)c * K + k];
  Y[i] = f2bf(s);
}

// ---------------------------------------------------------------------------
// Furthest point sampling: one block per batch, sequential over S samples
// ---------------------------------------------------------------------------
__global__ void k_fps(const float* __restrict__ xyz, int* __restrict__ fidx,
                      int N, int S) {
  int b = blockIdx.x, t = threadIdx.x;
  const float* p = xyz + (size_t)b * N * 3;
  __shared__ float dist[1024];
  __shared__ float rv[256];
  __shared__ int   ri[256];
  __shared__ int   s_far;
  for (int j = t; j < N; j += 256) dist[j] = 1e10f;
  if (t == 0) s_far = 0;
  __syncthreads();
  for (int it = 0; it < S; ++it) {
    int far = s_far;
    if (t == 0) fidx[b * S + it] = far;
    float cx = p[far * 3], cy = p[far * 3 + 1], cz = p[far * 3 + 2];
    float bv = -1.f; int bi = 0x7fffffff;
    for (int j = t; j < N; j += 256) {
      float dx = p[j * 3] - cx, dy = p[j * 3 + 1] - cy, dz = p[j * 3 + 2] - cz;
      float d = dx * dx + dy * dy + dz * dz;
      float nd = fminf(dist[j], d);
      dist[j] = nd;
      if (nd > bv) { bv = nd; bi = j; }
    }
    rv[t] = bv; ri[t] = bi;
    __syncthreads();
    for (int k = 128; k; k >>= 1) {
      if (t < k) {
        if (rv[t + k] > rv[t] || (rv[t + k] == rv[t] && ri[t + k] < ri[t])) {
          rv[t] = rv[t + k]; ri[t] = ri[t + k];
        }
      }
      __syncthreads();
    }
    if (t == 0) s_far = ri[0];
    __syncthreads();
  }
}

// kNN (k=32): one block per (b,s); also emits new_xyz
__global__ void k_knn(const float* __restrict__ xyz, const int* __restrict__ fidx,
                      float* __restrict__ new_xyz, int* __restrict__ kidx,
                      int N, int S) {
  int bs = blockIdx.x, t = threadIdx.x;
  int b = bs / S;
  const float* p = xyz + (size_t)b * N * 3;
  int ci = fidx[bs];
  float cx = p[ci * 3], cy = p[ci * 3 + 1], cz = p[ci * 3 + 2];
  if (t == 0) {
    new_xyz[(size_t)bs * 3]     = cx;
    new_xyz[(size_t)bs * 3 + 1] = cy;
    new_xyz[(size_t)bs * 3 + 2] = cz;
  }
  __shared__ float dist[1024];
  __shared__ float rv[256];
  __shared__ int   ri[256];
  for (int j = t; j < N; j += 256) {
    float dx = p[j * 3] - cx, dy = p[j * 3 + 1] - cy, dz = p[j * 3 + 2] - cz;
    dist[j] = dx * dx + dy * dy + dz * dz;
  }
  __syncthreads();
  for (int sel = 0; sel < 32; ++sel) {
    float bv = 1e30f; int bi = 0x7fffffff;
    for (int j = t; j < N; j += 256) {
      float d = dist[j];
      if (d < bv) { bv = d; bi = j; }
    }
    rv[t] = bv; ri[t] = bi;
    __syncthreads();
    for (int k = 128; k; k >>= 1) {
      if (t < k) {
        if (rv[t + k] < rv[t] || (rv[t + k] == rv[t] && ri[t + k] < ri[t])) {
          rv[t] = rv[t + k]; ri[t] = ri[t + k];
        }
      }
      __syncthreads();
    }
    if (t == 0) { kidx[(size_t)bs * 32 + sel] = ri[0]; dist[ri[0]] = 1e30f; }
    __syncthreads();
  }
}

// mean over k per (b,s,channel) of gathered [feat|xyz]
__global__ void k_gmean(const unsigned short* __restrict__ feat,
                        const float* __restrict__ xyz,
                        const int* __restrict__ kidx,
                        float* __restrict__ meanb, int N, int S, int C) {
  int CP = C + 3;
  int i = blockIdx.x * 256 + threadIdx.x;
  if (i >= 16 * S * CP) return;
  int c = i % CP; int bs = i / CP; int b = bs / S;
  float s = 0.f;
  for (int j = 0; j < 32; ++j) {
    int n = kidx[(size_t)bs * 32 + j];
    s += gval(feat, xyz, b, n, c, C, N);
  }
  meanb[i] = s * (1.f / 32.f);
}

// per-batch std (ddof=1) of (g - mean) over flattened S*k*(C+3)
__global__ void k_bstd(const unsigned short* __restrict__ feat,
                       const float* __restrict__ xyz,
                       const int* __restrict__ kidx,
                       const float* __restrict__ meanb,
                       float* __restrict__ stdb, int N, int S, int C) {
  int b = blockIdx.x, t = threadIdx.x;
  int CP = C + 3;
  size_t cnt = (size_t)S * 32 * CP;
  float acc = 0.f;
  for (size_t i = t; i < cnt; i += 256) {
    int c = (int)(i % (size_t)CP);
    size_t rem = i / (size_t)CP;
    int j = (int)(rem % 32);
    int s = (int)(rem / 32);
    int bs = b * S + s;
    int n = kidx[(size_t)bs * 32 + j];
    float v = gval(feat, xyz, b, n, c, C, N) - meanb[(size_t)bs * CP + c];
    acc += v * v;
  }
  __shared__ float red[256];
  red[t] = acc;
  __syncthreads();
  for (int k = 128; k; k >>= 1) {
    if (t < k) red[t] += red[t + k];
    __syncthreads();
  }
  if (t == 0) stdb[b] = sqrtf(red[0] / (float)(cnt - 1));
}

// write normalized grouped tensor + anchor concat into zero-padded bf16 G
__global__ void k_gwrite(const unsigned short* __restrict__ feat,
                         const float* __restrict__ xyz,
                         const int* __restrict__ kidx,
                         const int* __restrict__ fidx,
                         const float* __restrict__ meanb,
                         const float* __restrict__ stdb,
                         const float* __restrict__ alpha,
                         const float* __restrict__ beta,
                         unsigned short* __restrict__ G,
                         int N, int S, int C, int Kp) {
  size_t i = (size_t)blockIdx.x * 256 + threadIdx.x;
  size_t total = (size_t)16 * S * 32 * Kp;
  if (i >= total) return;
  int c = (int)(i % (size_t)Kp);
  size_t r = i / (size_t)Kp;
  int j = (int)(r % 32);
  int bs = (int)(r / 32);
  int b = bs / S;
  int CP = C + 3;
  unsigned short o = 0;
  if (c < CP) {
    int n = kidx[(size_t)bs * 32 + j];
    float v = gval(feat, xyz, b, n, c, C, N);
    float m = meanb[(size_t)bs * CP + c];
    o = f2bf(alpha[c] * ((v - m) / (stdb[b] + 1e-5f)) + beta[c]);
  } else if (c < 2 * C + 3) {
    int n = fidx[bs];
    o = feat[((size_t)b * N + n) * C + (c - CP)];   // anchor: raw bf16 copy
  }
  G[i] = o;
}

// max over k: H[bs,k,C] -> out[bs,C] (bf16); 8 channels per thread (b128 loads)
__global__ void k_maxk8(const unsigned short* __restrict__ H,
                        unsigned short* __restrict__ out, int BS, int C) {
  int cw = C / 8;
  size_t i = (size_t)blockIdx.x * 256 + threadIdx.x;
  if (i >= (size_t)BS * cw) return;
  size_t bs = i / (size_t)cw;
  int c0 = (int)(i % (size_t)cw) * 8;
  float m[8];
#pragma unroll
  for (int e = 0; e < 8; ++e) m[e] = -3.0e38f;
  for (int j = 0; j < 32; ++j) {
    uint4 hv = *(const uint4*)&H[((size_t)bs * 32 + j) * C + c0];
    const unsigned short* hp = (const unsigned short*)&hv;
#pragma unroll
    for (int e = 0; e < 8; ++e) m[e] = fmaxf(m[e], bf2f(hp[e]));
  }
  uint4 ov;
  unsigned short* op = (unsigned short*)&ov;
#pragma unroll
  for (int e = 0; e < 8; ++e) op[e] = f2bf(m[e]);
  *(uint4*)&out[bs * C + c0] = ov;
}

// global max over S: feat[b,S,C] -> v[b,C] (fp32)
__global__ void k_gmax(const unsigned short* __restrict__ feat,
                       float* __restrict__ v, int S, int C) {
  int i = blockIdx.x * 256 + threadIdx.x;
  if (i >= 16 * C) return;
  int c = i % C, b = i / C;
  float m = -3.0e38f;
  for (int s = 0; s < S; ++s)
    m = fmaxf(m, bf2f(feat[((size_t)b * S + s) * C + c]));
  v[i] = m;
}

// classifier fp32 linear (rows=16, tiny)
__global__ void k_cls_lin(const float* __restrict__ X, const float* __restrict__ W,
                          const float* __restrict__ b, float* __restrict__ Y,
                          int rows, int N, int K) {
  int i = blockIdx.x * 256 + threadIdx.x;
  if (i >= rows * N) return;
  int r = i / N, c = i % N;
  float s = b[c];
  for (int k = 0; k < K; ++k) s += X[(size_t)r * K + k] * W[(size_t)c * K + k];
  Y[i] = s;
}

// classifier BN (over 16 rows) + relu, in place; one thread per channel
__global__ void k_cls_bn(float* __restrict__ Y, const float* __restrict__ g,
                         const float* __restrict__ be, int rows, int C) {
  int c = blockIdx.x * 256 + threadIdx.x;
  if (c >= C) return;
  float s = 0.f, q = 0.f;
  for (int r = 0; r < rows; ++r) { float v = Y[r * C + c]; s += v; q += v * v; }
  float m = s / rows;
  float var = q / rows - m * m;
  float rs = rsqrtf(var + 1e-5f);
  for (int r = 0; r < rows; ++r) {
    float v = g[c] * (Y[r * C + c] - m) * rs + be[c];
    Y[r * C + c] = fmaxf(v, 0.f);
  }
}

// ---------------------------------------------------------------------------
// host driver
// ---------------------------------------------------------------------------
extern "C" void kernel_launch(void* const* d_in, const int* in_sizes, int n_in,
                              void* d_out, int out_size, void* d_ws, size_t ws_size,
                              hipStream_t stream) {
  (void)in_sizes; (void)n_in; (void)out_size; (void)ws_size;

  struct LinP { const float *w, *b, *g, *be; };
  int idx = 0;
  auto nxt = [&]() { return (const float*)d_in[idx++]; };
  auto nxtlin = [&]() { LinP l; l.w = nxt(); l.b = nxt(); l.g = nxt(); l.be = nxt(); return l; };

  // setup_inputs() order: x, then params in insertion order
  const float* x = nxt();
  LinP embed = nxtlin();
  struct StageP { const float *alpha, *beta; LinP transfer; LinP res[8]; };
  StageP st[4];
  for (int i = 0; i < 4; ++i) {
    st[i].alpha = nxt();
    st[i].beta  = nxt();
    st[i].transfer = nxtlin();
    for (int r = 0; r < 8; ++r) st[i].res[r] = nxtlin();  // pre0.c1,c2 pre1.c1,c2 pos0.c1,c2 pos1.c1,c2
  }
  LinP l1 = nxtlin(), l2 = nxtlin();
  const float* w3 = nxt();
  const float* b3 = nxt();

  const int Cin[4]  = {64, 128, 256, 512};
  const int Cout[4] = {128, 256, 512, 1024};
  const int Kp[4]   = {160, 288, 544, 1056};   // round_up(3+2*Cin, 32)

  // ---- workspace carve ----
  char* base = (char*)d_ws;
  size_t off = 0;
  auto carve = [&](size_t bytes) -> void* {
    void* p = base + off;
    off += (bytes + 255) & ~(size_t)255;
    return p;
  };
  float* XYZA  = (float*)carve((size_t)16 * 1024 * 3 * 4);
  float* XYZB  = (float*)carve((size_t)16 * 1024 * 3 * 4);
  unsigned short* FEAT = (unsigned short*)carve((size_t)16 * 65536 * 2);  // S*C == 65536 at every stage
  int*   FIDX  = (int*)carve((size_t)16 * 512 * 4);
  int*   KIDX  = (int*)carve((size_t)16 * 512 * 32 * 4);
  float* MEANB = (float*)carve((size_t)16 * 67072 * 4);   // max S*(C+3)
  float* STDB  = (float*)carve(16 * 4);
  float* STATS = (float*)carve(2 * 1024 * 4);
  float* VBUF  = (float*)carve(16 * 1024 * 4);
  float* V1    = (float*)carve(16 * 512 * 4);
  float* V2    = (float*)carve(16 * 256 * 4);
  unsigned short* WT[4];
  unsigned short* WR[4][8];
  for (int i = 0; i < 4; ++i) {
    WT[i] = (unsigned short*)carve((size_t)Cout[i] * Kp[i] * 2);
    for (int r = 0; r < 8; ++r)
      WR[i][r] = (unsigned short*)carve((size_t)Cout[i] * Cout[i] * 2);
  }
  unsigned short* BUF_G = (unsigned short*)carve((size_t)16 * 512 * 32 * 160 * 2); // 84 MB
  unsigned short* BUF_A = (unsigned short*)carve((size_t)16 * 512 * 32 * 128 * 2); // 67 MB
  unsigned short* BUF_B = (unsigned short*)carve((size_t)16 * 512 * 32 * 128 * 2); // 67 MB

  auto blocks = [](size_t tot) { return (unsigned)((tot + 255) / 256); };

  // ---- convert weights to padded bf16 ----
  for (int i = 0; i < 4; ++i) {
    int cin = 3 + 2 * Cin[i];
    k_convw<<<blocks((size_t)Cout[i] * Kp[i]), 256, 0, stream>>>(
        st[i].transfer.w, WT[i], Cout[i], cin, Kp[i]);
    for (int r = 0; r < 8; ++r)
      k_convw<<<blocks((size_t)Cout[i] * Cout[i]), 256, 0, stream>>>(
          st[i].res[r].w, WR[i][r], Cout[i], Cout[i], Cout[i]);
  }

  auto gemm = [&](const unsigned short* Xb, int ldx, const unsigned short* Wb, int ldw,
                  const float* bias, unsigned short* Yb, int ldy, int M, int Nn, int Kk) {
    dim3 grid(Nn / 128, M / 128);
    k_gemm<<<grid, 256, 0, stream>>>(Xb, ldx, Wb, ldw, bias, Yb, ldy, M, Nn, Kk);
  };
  auto bn = [&](const unsigned short* Yb, const unsigned short* Rb, unsigned short* Ob,
                const float* g, const float* be, int rows, int Cc) {
    k_bn_stats<<<Cc, 256, 0, stream>>>(Yb, Cc, rows, STATS);
    size_t tot8 = (size_t)rows * Cc / 8;
    k_bnapply8<<<blocks(tot8), 256, 0, stream>>>(Yb, Rb, Ob, STATS, g, be, tot8, Cc);
  };

  // ---- embedding: x[B,3,N] -> xyz[B,N,3] -> feat bf16 [B*N, 64] ----
  k_xpose<<<blocks(16 * 1024), 256, 0, stream>>>(x, XYZA, 1024);
  int rows0 = 16 * 1024;
  k_lin_small<<<blocks((size_t)rows0 * 64), 256, 0, stream>>>(
      XYZA, embed.w, embed.b, BUF_A, rows0, 64, 3);
  bn(BUF_A, nullptr, FEAT, embed.g, embed.be, rows0, 64);

  // ---- stages ----
  float* xyz_cur = XYZA;
  float* xyz_nxt = XYZB;
  int N = 1024, C = 64;
  for (int i = 0; i < 4; ++i) {
    int S = N / 2, Co = Cout[i], kp = Kp[i];
    k_fps<<<16, 256, 0, stream>>>(xyz_cur, FIDX, N, S);
    k_knn<<<16 * S, 256, 0, stream>>>(xyz_cur, FIDX, xyz_nxt, KIDX, N, S);
    k_gmean<<<blocks((size_t)16 * S * (C + 3)), 256, 0, stream>>>(
        FEAT, xyz_cur, KIDX, MEANB, N, S, C);
    k_bstd<<<16, 256, 0, stream>>>(FEAT, xyz_cur, KIDX, MEANB, STDB, N, S, C);
    k_gwrite<<<blocks((size_t)16 * S * 32 * kp), 256, 0, stream>>>(
        FEAT, xyz_cur, KIDX, FIDX, MEANB, STDB, st[i].alpha, st[i].beta,
        BUF_G, N, S, C, kp);

    int rows = 16 * S * 32;
    // transfer CBR
    gemm(BUF_G, kp, WT[i], kp, st[i].transfer.b, BUF_A, Co, rows, Co, kp);
    bn(BUF_A, nullptr, BUF_A, st[i].transfer.g, st[i].transfer.be, rows, Co);
    // pre residual blocks
    for (int rb = 0; rb < 2; ++rb) {
      const LinP& c1 = st[i].res[rb * 2];
      const LinP& c2 = st[i].res[rb * 2 + 1];
      gemm(BUF_A, Co, WR[i][rb * 2], Co, c1.b, BUF_B, Co, rows, Co, Co);
      bn(BUF_B, nullptr, BUF_B, c1.g, c1.be, rows, Co);
      gemm(BUF_B, Co, WR[i][rb * 2 + 1], Co, c2.b, BUF_G, Co, rows, Co, Co);
      bn(BUF_G, BUF_A, BUF_A, c2.g, c2.be, rows, Co);     // + residual, relu
    }
    // max over k
    int rows2 = 16 * S;
    k_maxk8<<<blocks((size_t)rows2 * Co / 8), 256, 0, stream>>>(BUF_A, FEAT, rows2, Co);
    // pos residual blocks
    for (int rb = 2; rb < 4; ++rb) {
      const LinP& c1 = st[i].res[rb * 2];
      const LinP& c2 = st[i].res[rb * 2 + 1];
      gemm(FEAT, Co, WR[i][rb * 2], Co, c1.b, BUF_B, Co, rows2, Co, Co);
      bn(BUF_B, nullptr, BUF_B, c1.g, c1.be, rows2, Co);
      gemm(BUF_B, Co, WR[i][rb * 2 + 1], Co, c2.b, BUF_G, Co, rows2, Co, Co);
      bn(BUF_G, FEAT, FEAT, c2.g, c2.be, rows2, Co);
    }
    float* tmp = xyz_cur; xyz_cur = xyz_nxt; xyz_nxt = tmp;
    N = S; C = Co;
  }

  // ---- classifier ----
  k_gmax<<<blocks(16 * 1024), 256, 0, stream>>>(FEAT, VBUF, N, C);
  k_cls_lin<<<blocks(16 * 512), 256, 0, stream>>>(VBUF, l1.w, l1.b, V1, 16, 512, 1024);
  k_cls_bn<<<blocks(512), 256, 0, stream>>>(V1, l1.g, l1.be, 16, 512);
  k_cls_lin<<<blocks(16 * 256), 256, 0, stream>>>(V1, l2.w, l2.b, V2, 16, 256, 512);
  k_cls_bn<<<blocks(256), 256, 0, stream>>>(V2, l2.g, l2.be, 16, 256);
  k_cls_lin<<<blocks(16 * 40), 256, 0, stream>>>(V2, w3, b3, (float*)d_out, 16, 40, 256);
}
